// FactorizedTransition_33483565040298
// MI455X (gfx1250) — matrix operand order repaired and checked by reading, not compiled, not run on hardware
//
#include <hip/hip_runtime.h>

#define B_ 4
#define S_ 4096
#define E_ 128
#define H_ 128
#define SSTRIDE 4112   // 4096 + 16 pad -> lane-halves hit disjoint LDS bank windows

typedef __attribute__((ext_vector_type(16))) _Float16 v16h;
typedef __attribute__((ext_vector_type(8)))  float    v8f;

// Workspace layout (halves):
//   emb16 : [S][E]            row-major, K(=E) contiguous
//   featT : [B][2][H][E]      per-action features, pre-transposed so K(=E) contiguous
//   proj  : [B][2][S][H]      qk=0 -> Q (query_features), qk=1 -> K (key_features)

// A-fragment (16x32 f16, MxK): lane m=l&15, h=l>>4 holds contiguous K = k0+16h .. k0+16h+15
__device__ __forceinline__ v16h load_a_frag(const _Float16* __restrict__ base,
                                            int row0, int ld, int k0, int lane) {
  const int m = lane & 15, h = lane >> 4;
  const _Float16* p = base + (size_t)(row0 + m) * ld + k0 + 16 * h;
  v16h r;
  ((int4*)&r)[0] = *(const int4*)(p);
  ((int4*)&r)[1] = *(const int4*)(p + 8);
  return r;
}

// B-fragment (32x16 f16, KxN) from row-major B^T [N][K]: lane n=l&15, h=l>>4
// holds contiguous K = k0+16h .. k0+16h+15
__device__ __forceinline__ v16h load_b_frag(const _Float16* __restrict__ base,
                                            int col0, int ld, int k0, int lane) {
  const int n = lane & 15, h = lane >> 4;
  const _Float16* p = base + (size_t)(col0 + n) * ld + k0 + 16 * h;
  v16h r;
  ((int4*)&r)[0] = *(const int4*)(p);
  ((int4*)&r)[1] = *(const int4*)(p + 8);
  return r;
}

__global__ void zero_out_kernel(float* __restrict__ out) {
  out[blockIdx.x * 256 + threadIdx.x] = 0.f;
}

__global__ void cvt_emb_kernel(const float* __restrict__ src, _Float16* __restrict__ dst) {
  const int i = blockIdx.x * 256 + threadIdx.x;  // S_*E_ elements
  dst[i] = (_Float16)src[i];
}

// featT[b][qk][n][k] = feat[action[b]][k][n]  (qk=0: query_features, 1: key_features)
__global__ void cvt_feat_kernel(const float* __restrict__ keyf,
                                const float* __restrict__ qryf,
                                const int* __restrict__ action,
                                _Float16* __restrict__ featT) {
  const int i = blockIdx.x * 256 + threadIdx.x;  // B_*2*H_*E_ elements
  const int k  = i % E_;
  int t        = i / E_;
  const int nn = t % H_;  t /= H_;
  const int qk = t & 1;
  const int b  = t >> 1;
  const float* src = qk ? keyf : qryf;
  const int a = action[b];
  featT[i] = (_Float16)src[((size_t)a * E_ + k) * H_ + nn];
}

// proj[b][qk][s][h] = emb16[s][:] . featT[b][qk][h][:]   (M=4096, N=128, K=128)
__global__ __launch_bounds__(256) void proj_kernel(const _Float16* __restrict__ emb16,
                                                   const _Float16* __restrict__ featT,
                                                   _Float16* __restrict__ proj) {
  const int tid = threadIdx.x, lane = tid & 31, wave = tid >> 5;
  const int rb = blockIdx.x, qk = blockIdx.y, b = blockIdx.z;
  const int row0 = rb * 128 + wave * 16;
  const _Float16* fT = featT + (size_t)(b * 2 + qk) * H_ * E_;
  _Float16* P = proj + (size_t)(b * 2 + qk) * S_ * H_;

  const v16h a0 = load_a_frag(emb16, row0, E_, 0,  lane);
  const v16h a1 = load_a_frag(emb16, row0, E_, 32, lane);
  const v16h a2 = load_a_frag(emb16, row0, E_, 64, lane);
  const v16h a3 = load_a_frag(emb16, row0, E_, 96, lane);
  const int n = lane & 15, hh = lane >> 4;

  for (int nt = 0; nt < 8; ++nt) {
    const int n0 = nt * 16;
    v8f c = {};
    v16h bf = load_b_frag(fT, n0, E_, 0, lane);
    c = __builtin_amdgcn_wmma_f32_16x16x32_f16(false, a0, false, bf, (short)0, c, false, false);
    bf = load_b_frag(fT, n0, E_, 32, lane);
    c = __builtin_amdgcn_wmma_f32_16x16x32_f16(false, a1, false, bf, (short)0, c, false, false);
    bf = load_b_frag(fT, n0, E_, 64, lane);
    c = __builtin_amdgcn_wmma_f32_16x16x32_f16(false, a2, false, bf, (short)0, c, false, false);
    bf = load_b_frag(fT, n0, E_, 96, lane);
    c = __builtin_amdgcn_wmma_f32_16x16x32_f16(false, a3, false, bf, (short)0, c, false, false);
#pragma unroll
    for (int v = 0; v < 8; ++v)
      P[(size_t)(row0 + v + 8 * hh) * H_ + n0 + n] = (_Float16)c[v];
  }
}

// One WG = 16 rows of s, full 4096 columns of t, scores kept in LDS.
__global__ __launch_bounds__(256) void attn_kernel(const _Float16* __restrict__ proj,
                                                   const float* __restrict__ belief,
                                                   float* __restrict__ out) {
  extern __shared__ float smem[];
  float* sc  = smem;                 // [16][SSTRIDE] scores / probs
  float* red = smem + 16 * SSTRIDE;  // 256 reduction partials
  float* rw  = red + 256;            // 16: row max, then row weight

  const int tid = threadIdx.x, lane = tid & 31, wave = tid >> 5;
  const int row0 = blockIdx.x * 16;
  const int b    = blockIdx.y;

  const _Float16* Qb = proj + (size_t)(b * 2 + 0) * S_ * H_;
  const _Float16* Kb = proj + (size_t)(b * 2 + 1) * S_ * H_;

  const v16h a0 = load_a_frag(Qb, row0, H_, 0,  lane);
  const v16h a1 = load_a_frag(Qb, row0, H_, 32, lane);
  const v16h a2 = load_a_frag(Qb, row0, H_, 64, lane);
  const v16h a3 = load_a_frag(Qb, row0, H_, 96, lane);
  const int n = lane & 15, hh = lane >> 4;

  // scores[16][col0..col0+15] per tile; wave w owns columns [512w, 512w+512)
  for (int nt = 0; nt < 32; ++nt) {
    const int col0 = wave * 512 + nt * 16;
    v8f c = {};
    v16h bf = load_b_frag(Kb, col0, H_, 0, lane);
    c = __builtin_amdgcn_wmma_f32_16x16x32_f16(false, a0, false, bf, (short)0, c, false, false);
    bf = load_b_frag(Kb, col0, H_, 32, lane);
    c = __builtin_amdgcn_wmma_f32_16x16x32_f16(false, a1, false, bf, (short)0, c, false, false);
    bf = load_b_frag(Kb, col0, H_, 64, lane);
    c = __builtin_amdgcn_wmma_f32_16x16x32_f16(false, a2, false, bf, (short)0, c, false, false);
    bf = load_b_frag(Kb, col0, H_, 96, lane);
    c = __builtin_amdgcn_wmma_f32_16x16x32_f16(false, a3, false, bf, (short)0, c, false, false);
#pragma unroll
    for (int v = 0; v < 8; ++v)
      sc[(v + 8 * hh) * SSTRIDE + col0 + n] = c[v];
  }
  __syncthreads();

  // ---- row max (16 threads per row, then 16-way combine) ----
  const int r = tid >> 4, j0 = tid & 15;
  float mx = -3.4e38f;
  for (int j = j0; j < S_; j += 16) mx = fmaxf(mx, sc[r * SSTRIDE + j]);
  red[tid] = mx;
  __syncthreads();
  if (tid < 16) {
    float mm = red[tid * 16];
#pragma unroll
    for (int i = 1; i < 16; ++i) mm = fmaxf(mm, red[tid * 16 + i]);
    rw[tid] = mm;
  }
  __syncthreads();

  // ---- exp in place + row sum; then w_r = belief_r / Z_r ----
  const float mr = rw[r];
  float sum = 0.f;
  for (int j = j0; j < S_; j += 16) {
    const float p = __expf(sc[r * SSTRIDE + j] - mr);
    sc[r * SSTRIDE + j] = p;
    sum += p;
  }
  red[tid] = sum;
  __syncthreads();
  if (tid < 16) {
    float z = 0.f;
#pragma unroll
    for (int i = 0; i < 16; ++i) z += red[tid * 16 + i];
    rw[tid] = belief[b * S_ + row0 + tid] / z;
  }
  __syncthreads();

  // ---- out[t] += sum_r w_r * p[r][t]  (coalesced columns, f32 atomics) ----
#pragma unroll
  for (int i = 0; i < 16; ++i) {
    const int cidx = tid + 256 * i;
    float acc = 0.f;
#pragma unroll
    for (int r2 = 0; r2 < 16; ++r2) acc += rw[r2] * sc[r2 * SSTRIDE + cidx];
    atomicAdd(&out[b * S_ + cidx], acc);
  }
}

extern "C" void kernel_launch(void* const* d_in, const int* in_sizes, int n_in,
                              void* d_out, int out_size, void* d_ws, size_t ws_size,
                              hipStream_t stream) {
  const float* belief         = (const float*)d_in[0];
  const int*   action         = (const int*)d_in[1];
  const float* state_emb      = (const float*)d_in[2];
  const float* key_features   = (const float*)d_in[3];
  const float* query_features = (const float*)d_in[4];
  float* out = (float*)d_out;

  _Float16* emb16 = (_Float16*)d_ws;                       // 1 MB
  _Float16* featT = emb16 + (size_t)S_ * E_;               // 256 KB
  _Float16* proj  = featT + (size_t)B_ * 2 * H_ * E_;      // 8 MB

  zero_out_kernel<<<(B_ * S_) / 256, 256, 0, stream>>>(out);
  cvt_emb_kernel<<<(S_ * E_) / 256, 256, 0, stream>>>(state_emb, emb16);
  cvt_feat_kernel<<<(B_ * 2 * H_ * E_) / 256, 256, 0, stream>>>(key_features, query_features,
                                                                action, featT);
  proj_kernel<<<dim3(S_ / 128, 2, B_), 256, 0, stream>>>(emb16, featT, proj);

  const size_t shmem = (size_t)(16 * SSTRIDE + 256 + 16) * sizeof(float);  // ~258 KB < 320 KB
  attn_kernel<<<dim3(S_ / 16, B_), 256, shmem, stream>>>(proj, belief, out);
}
